// CrystalGATEncoder_79637283602848
// MI455X (gfx1250) — matrix operand order repaired and checked by reading, not compiled, not run on hardware
//
#include <hip/hip_runtime.h>
#include <hip/hip_bf16.h>

typedef __bf16 bf16;
typedef __attribute__((ext_vector_type(16))) __bf16 v16bf;
typedef __attribute__((ext_vector_type(8)))  __bf16 v8bf;
typedef __attribute__((ext_vector_type(8)))  float  v8f;

#define TPB 256

__device__ __forceinline__ float siluf(float x){ return x / (1.0f + __expf(-x)); }

__device__ __forceinline__ void atomicMaxF(float* addr, float val){
  int* ia = (int*)addr;
  int old = __float_as_int(*addr);
  while (__int_as_float(old) < val){
    int assumed = old;
    old = atomicCAS(ia, assumed, __float_as_int(val));
    if (old == assumed) break;
  }
}

// ------------------------------ fills / converts ------------------------------
__global__ void fill_f32(float* p, float v, long long n){
  long long t = blockIdx.x*(long long)blockDim.x + threadIdx.x;
  if (t < n) p[t] = v;
}

__global__ void convert_pad_bf16(const float* __restrict__ A, bf16* __restrict__ B,
                                 int N, int Mp, int K, int Kpad){
  long long t = blockIdx.x*(long long)blockDim.x + threadIdx.x;
  if (t >= (long long)Mp*Kpad) return;
  int n = (int)(t / Kpad), k = (int)(t % Kpad);
  float v = (n < N && k < K) ? A[(size_t)n*K + k] : 0.0f;
  B[t] = (bf16)v;
}

// Pack fp32 W[K,Nout] into WMMA B-operand tiles (bf16):
// P[((kt*NT+nt)*32+lane)*16 + j]:
//   lane<16 : B[kt*32 +      j][nt*16 + lane]
//   lane>=16: B[kt*32 + 16 + j][nt*16 + lane-16]
__global__ void pack_weight(const float* __restrict__ W, bf16* __restrict__ P,
                            int K, int Kpad, int Nout){
  long long total = (long long)(Kpad >> 5) * (Nout >> 4) * 512;
  long long t = blockIdx.x*(long long)blockDim.x + threadIdx.x;
  if (t >= total) return;
  int j    = (int)(t & 15);
  int lane = (int)((t >> 4) & 31);
  long long tile = t >> 9;
  int NT = Nout >> 4;
  int nt = (int)(tile % NT);
  int kt = (int)(tile / NT);
  int k = kt*32 + ((lane >= 16) ? 16 : 0) + j;
  int n = nt*16 + (lane & 15);
  float v = (k < K) ? W[(size_t)k*Nout + n] : 0.0f;
  P[t] = (bf16)v;
}

// ------------------------------ WMMA GEMM ------------------------------
// C[Nrows,Nout](f32) = A[Mp,Kpad](bf16, row-major, K zero-padded) * Bpack (+bias)
// Register-blocked 2 M-tiles x 2 N-tiles per wave: 4 independent
// v_wmma_f32_16x16x32_bf16 chains per K-step, 2 b128 loads per WMMA.
// Block = 8 waves -> 16 N-tiles (256 cols) x 32 rows per block.
__device__ __forceinline__ v16bf load_a_frag(const bf16* arow, int k0, int half){
  v8bf lo = *(const v8bf*)(arow + k0 + half*16);
  v8bf hi = *(const v8bf*)(arow + k0 + half*16 + 8);
  v16bf a;
#pragma unroll
  for (int i = 0; i < 8; ++i){ a[i] = lo[i]; a[i+8] = hi[i]; }
  return a;
}

__global__ void gemm_wmma_bf16(const bf16* __restrict__ A, const bf16* __restrict__ Bp,
                               float* __restrict__ C, const float* __restrict__ bias,
                               int Nrows, int Kpad, int Nout, int useBias){
  int lane = threadIdx.x & 31;
  int wave = threadIdx.x >> 5;
  int NT = Nout >> 4;
  int nt0 = blockIdx.x * 16 + wave * 2;     // first of 2 N-tiles for this wave
  if (nt0 >= NT) return;
  bool n1ok = (nt0 + 1) < NT;
  int nt1 = n1ok ? (nt0 + 1) : nt0;         // safe dummy when out of range
  int m0 = blockIdx.y << 5;                 // 32 rows: two 16-row M-tiles
  int half = lane >> 4;                     // 0: lanes 0-15, 1: lanes 16-31
  int r = lane & 15;

  const bf16* arow0 = A + (size_t)(m0 +      r) * Kpad;   // row r; element s holds k = 16*half + s (contiguous, same as Bpack)
  const bf16* arow1 = A + (size_t)(m0 + 16 + r) * Kpad;
  v8f acc00 = {0.f,0.f,0.f,0.f,0.f,0.f,0.f,0.f};
  v8f acc01 = acc00, acc10 = acc00, acc11 = acc00;

  int KT = Kpad >> 5;
  for (int kt = 0; kt < KT; ++kt){
    int k0 = kt << 5;
    v16bf a0 = load_a_frag(arow0, k0, half);
    v16bf a1 = load_a_frag(arow1, k0, half);
    v16bf b0 = *(const v16bf*)(Bp + ((((size_t)kt*NT + nt0)*32) + lane)*16);
    v16bf b1 = *(const v16bf*)(Bp + ((((size_t)kt*NT + nt1)*32) + lane)*16);
    acc00 = __builtin_amdgcn_wmma_f32_16x16x32_bf16(false, a0, false, b0, (short)0, acc00, false, false);
    acc01 = __builtin_amdgcn_wmma_f32_16x16x32_bf16(false, a0, false, b1, (short)0, acc01, false, false);
    acc10 = __builtin_amdgcn_wmma_f32_16x16x32_bf16(false, a1, false, b0, (short)0, acc10, false, false);
    acc11 = __builtin_amdgcn_wmma_f32_16x16x32_bf16(false, a1, false, b1, (short)0, acc11, false, false);
  }

  // ISA C layout: lane<16 -> rows m..m+7 col=lane ; lane>=16 -> rows m+8..m+15 col=lane-16
  int col0 = (nt0 << 4) + r;
  int col1 = (nt1 << 4) + r;
  int rb0 = m0 + half*8;
  int rb1 = m0 + 16 + half*8;
  float bs0 = useBias ? bias[col0] : 0.0f;
  float bs1 = useBias ? bias[col1] : 0.0f;
#pragma unroll
  for (int i = 0; i < 8; ++i){
    int rr0 = rb0 + i, rr1 = rb1 + i;
    if (rr0 < Nrows){
      C[(size_t)rr0*Nout + col0] = acc00[i] + bs0;
      if (n1ok) C[(size_t)rr0*Nout + col1] = acc01[i] + bs1;
    }
    if (rr1 < Nrows){
      C[(size_t)rr1*Nout + col0] = acc10[i] + bs0;
      if (n1ok) C[(size_t)rr1*Nout + col1] = acc11[i] + bs1;
    }
  }
}

// ------------------------------ small dense ops ------------------------------
__global__ void edge_mlp_kernel(const float* __restrict__ efin,
                                const float* __restrict__ W1, const float* __restrict__ b1,
                                const float* __restrict__ W2, const float* __restrict__ b2,
                                float* __restrict__ efout, int E){
  int e = blockIdx.x*blockDim.x + threadIdx.x;
  if (e >= E) return;
  float x0 = efin[e*3+0], x1 = efin[e*3+1], x2 = efin[e*3+2];
  float h[6];
#pragma unroll
  for (int j = 0; j < 6; ++j){
    float z = x0*W1[0*6+j] + x1*W1[1*6+j] + x2*W1[2*6+j] + b1[j];
    h[j] = siluf(z);
  }
#pragma unroll
  for (int c = 0; c < 3; ++c){
    float z = b2[c];
#pragma unroll
    for (int j = 0; j < 6; ++j) z += h[j]*W2[j*3+c];
    efout[e*3+c] = z;
  }
}

__global__ void xp0_kernel(const float* __restrict__ x, const float* __restrict__ W,
                           const float* __restrict__ b, float* __restrict__ xp, int N){
  int t = blockIdx.x*blockDim.x + threadIdx.x;
  if (t >= N*4) return;
  int n = t >> 2, o = t & 3;
  float s = b[o];
#pragma unroll
  for (int i = 0; i < 4; ++i) s += x[n*4+i]*W[i*4+o];
  xp[t] = s;
}

// ------------------------------ graph scatter / attention ------------------------------
__global__ void deg_ef_scatter(const float* __restrict__ ef, const int* __restrict__ src,
                               const int* __restrict__ dst, float* __restrict__ deg,
                               float* __restrict__ efacc, int E){
  int e = blockIdx.x*blockDim.x + threadIdx.x;
  if (e >= E) return;
  int s = src[e], d = dst[e];
  atomicAdd(&deg[s], 1.0f);
  atomicAdd(&deg[d], 1.0f);
#pragma unroll
  for (int c = 0; c < 3; ++c){
    float v = ef[e*3+c];
    atomicAdd(&efacc[s*3+c], v);
    atomicAdd(&efacc[d*3+c], v);
  }
}

// temp[:, :fin] = (outdeg+indeg)*xp ; temp[:, fin:fin+3] = scattered edge feats ; zero pad
__global__ void build_tempb(const float* __restrict__ xp, const float* __restrict__ deg,
                            const float* __restrict__ efacc, bf16* __restrict__ tempb,
                            int N, int Mp, int fin, int Kpad){
  long long t = blockIdx.x*(long long)blockDim.x + threadIdx.x;
  if (t >= (long long)Mp*Kpad) return;
  int n = (int)(t / Kpad), k = (int)(t % Kpad);
  float v = 0.0f;
  if (n < N){
    if (k < fin)            v = deg[n] * xp[(size_t)n*fin + k];
    else if (k < fin + 3)   v = efacc[(size_t)n*3 + (k - fin)];
  }
  tempb[t] = (bf16)v;
}

__global__ void el_er_kernel(const float* __restrict__ h, const float* __restrict__ al,
                             const float* __restrict__ ar, float* __restrict__ el,
                             float* __restrict__ er, int N, int F, int fout){
  int t = blockIdx.x*blockDim.x + threadIdx.x;
  if (t >= N*4) return;
  int n = t >> 2, hd = t & 3;
  const float* hp  = h  + (size_t)n*F + hd*fout;
  const float* alp = al + hd*fout;
  const float* arp = ar + hd*fout;
  float sl = 0.f, sr = 0.f;
  for (int f = 0; f < fout; ++f){ float v = hp[f]; sl += v*alp[f]; sr += v*arp[f]; }
  el[t] = sl; er[t] = sr;
}

__global__ void edge_logits(const float* __restrict__ el, const float* __restrict__ er,
                            const int* __restrict__ src, const int* __restrict__ dst,
                            float* __restrict__ ev, float* __restrict__ mx, int E){
  long long t = blockIdx.x*(long long)blockDim.x + threadIdx.x;
  if (t >= (long long)E*4) return;
  int e = (int)(t >> 2), hd = (int)(t & 3);
  float v = el[src[e]*4+hd] + er[dst[e]*4+hd];
  v = (v > 0.f) ? v : 0.2f*v;       // leaky_relu 0.2
  ev[t] = v;
  atomicMaxF(&mx[dst[e]*4+hd], v);
}

__global__ void edge_exp(float* __restrict__ ev, const int* __restrict__ dst,
                         const float* __restrict__ mx, float* __restrict__ ssum, int E){
  long long t = blockIdx.x*(long long)blockDim.x + threadIdx.x;
  if (t >= (long long)E*4) return;
  int e = (int)(t >> 2), hd = (int)(t & 3);
  float v = expf(ev[t] - mx[dst[e]*4+hd]);
  ev[t] = v;
  atomicAdd(&ssum[dst[e]*4+hd], v);
}

__global__ void edge_norm(float* __restrict__ ev, const int* __restrict__ dst,
                          const float* __restrict__ ssum, int E){
  long long t = blockIdx.x*(long long)blockDim.x + threadIdx.x;
  if (t >= (long long)E*4) return;
  int e = (int)(t >> 2), hd = (int)(t & 3);
  float s = ssum[dst[e]*4+hd];
  ev[t] = ev[t] / ((s == 0.f) ? 1.f : s);
}

__global__ void gat_scatter(const float* __restrict__ h, const float* __restrict__ a,
                            const int* __restrict__ src, const int* __restrict__ dst,
                            float* __restrict__ acc, long long total, int F, int fout){
  long long t = blockIdx.x*(long long)blockDim.x + threadIdx.x;
  if (t >= total) return;
  int e = (int)(t / F), f = (int)(t % F);
  int hd = f / fout;
  atomicAdd(&acc[(size_t)dst[e]*F + f], h[(size_t)src[e]*F + f] * a[(size_t)e*4 + hd]);
}

// ------------------------------ LayerNorm + SiLU epilogue ------------------------------
__global__ void ln_silu_kernel(const float* __restrict__ acc, const float* __restrict__ gat_b,
                               const float* __restrict__ g, const float* __restrict__ b,
                               float* __restrict__ out_f32, bf16* __restrict__ out_bf,
                               int F){
  int n = blockIdx.x;
  __shared__ float red[TPB];
  float s = 0.f, ss = 0.f;
  for (int f = threadIdx.x; f < F; f += blockDim.x){
    float v = acc[(size_t)n*F + f] + gat_b[f];
    s += v; ss += v*v;
  }
  red[threadIdx.x] = s; __syncthreads();
  for (int st = TPB/2; st > 0; st >>= 1){
    if (threadIdx.x < st) red[threadIdx.x] += red[threadIdx.x + st];
    __syncthreads();
  }
  float mean = red[0] / (float)F; __syncthreads();
  red[threadIdx.x] = ss; __syncthreads();
  for (int st = TPB/2; st > 0; st >>= 1){
    if (threadIdx.x < st) red[threadIdx.x] += red[threadIdx.x + st];
    __syncthreads();
  }
  float var = red[0] / (float)F - mean*mean;
  float rstd = rsqrtf(var + 1e-5f);
  for (int f = threadIdx.x; f < F; f += blockDim.x){
    float v = acc[(size_t)n*F + f] + gat_b[f];
    float y = (v - mean)*rstd*g[f] + b[f];
    float sl = siluf(y);
    if (out_f32) out_f32[(size_t)n*F + f] = sl;
    out_bf[(size_t)n*F + f] = (bf16)sl;
  }
}

// ------------------------------ pooling / heads ------------------------------
__global__ void pool_gate_kernel(const float* __restrict__ gate1, const float* __restrict__ W2,
                                 const float* __restrict__ b2, float* __restrict__ gate, int N){
  int n = blockIdx.x*blockDim.x + threadIdx.x;
  if (n >= N) return;
  float s = 0.f;
  for (int k = 0; k < 128; ++k){
    float v = siluf(gate1[(size_t)n*128 + k]);
    s += v * W2[k];
  }
  gate[n] = s + b2[0];
}

__global__ void graph_max(const float* __restrict__ gate, const int* __restrict__ gid,
                          float* __restrict__ gmax, int N){
  int n = blockIdx.x*blockDim.x + threadIdx.x;
  if (n >= N) return;
  atomicMaxF(&gmax[gid[n]], gate[n]);
}

__global__ void graph_expsum(const float* __restrict__ gate, const int* __restrict__ gid,
                             const float* __restrict__ gmax, float* __restrict__ wv,
                             float* __restrict__ gsum, int N){
  int n = blockIdx.x*blockDim.x + threadIdx.x;
  if (n >= N) return;
  int g = gid[n];
  float v = expf(gate[n] - gmax[g]);
  wv[n] = v;
  atomicAdd(&gsum[g], v);
}

__global__ void graph_emb_scatter(const float* __restrict__ h, const float* __restrict__ wv,
                                  const float* __restrict__ gsum, const int* __restrict__ gid,
                                  float* __restrict__ emb, int N){
  long long t = blockIdx.x*(long long)blockDim.x + threadIdx.x;
  if (t >= (long long)N*1024) return;
  int n = (int)(t >> 10), f = (int)(t & 1023);
  int g = gid[n];
  float s = gsum[g]; s = (s == 0.f) ? 1.f : s;
  atomicAdd(&emb[(size_t)g*1024 + f], (wv[n]/s) * h[(size_t)n*1024 + f]);
}

__global__ void mu_lv_kernel(const float* __restrict__ emb,
                             const float* __restrict__ muW, const float* __restrict__ mub,
                             const float* __restrict__ lvW, const float* __restrict__ lvb,
                             float* __restrict__ out, int G){
  int t = blockIdx.x*blockDim.x + threadIdx.x;
  if (t >= G*64) return;
  int g = t / 64, j = t % 64;
  float m = 0.f, l = 0.f;
  const float* e = emb + (size_t)g*1024;
  for (int k = 0; k < 1024; ++k){ float v = e[k]; m += v*muW[k*64+j]; l += v*lvW[k*64+j]; }
  m += mub[j]; l += lvb[j];
  l = fminf(10.f, fmaxf(-10.f, l));
  out[g*64 + j] = m;
  out[G*64 + g*64 + j] = l;
}

// ------------------------------ host driver ------------------------------
static inline unsigned nblk(long long total){ return (unsigned)((total + TPB - 1) / TPB); }

extern "C" void kernel_launch(void* const* d_in, const int* in_sizes, int n_in,
                              void* d_out, int out_size, void* d_ws, size_t ws_size,
                              hipStream_t stream){
  const float* node_feats = (const float*)d_in[0];
  const float* edge_feats = (const float*)d_in[1];
  const int*   src        = (const int*)d_in[2];
  const int*   dst        = (const int*)d_in[3];
  const int*   gid        = (const int*)d_in[4];
  const int N = in_sizes[0] / 4;
  const int E = in_sizes[2];
  const int G = 16;                       // reference constant (num_graphs)
  const int Mp = (N + 31) & ~31;          // 32-row M-blocks for the 2x2 register-blocked GEMM

  auto P = [&](int li, int k)->const float*{ return (const float*)d_in[6 + 14*li + k]; };
  // param indices: 0 em_W1,1 em_b1,2 em_W2,3 em_b2,4 np_W,5 np_b,6 fc_W,7 attn_l,
  //                8 attn_r,9 gat_b,10 res_W,11 res_b,12 ln_g,13 ln_b
  const float* pool_W1 = (const float*)d_in[48];
  const float* pool_b1 = (const float*)d_in[49];
  const float* pool_W2 = (const float*)d_in[50];
  const float* pool_b2 = (const float*)d_in[51];
  const float* mu_W    = (const float*)d_in[52];
  const float* mu_b    = (const float*)d_in[53];
  const float* lv_W    = (const float*)d_in[54];
  const float* lv_b    = (const float*)d_in[55];

  const int finA[3]  = {4, 256, 512};
  const int foutA[3] = {64, 128, 256};
  const int FA[3]    = {256, 512, 1024};
  const int KresA[3] = {32, 256, 512};     // roundup(fin,32)
  const int KfcA[3]  = {32, 288, 544};     // roundup(fin+3,32)

  // bump allocator over d_ws
  char* base = (char*)d_ws; size_t off = 0;
  auto alloc = [&](size_t bytes)->void*{
    void* p = base + off; off = (off + bytes + 255) & ~(size_t)255; return p;
  };
  bf16*  hbf   = (bf16*)alloc((size_t)Mp*1024*2);
  bf16*  xbf   = (bf16*)alloc((size_t)Mp*32*2);
  bf16*  tempb = (bf16*)alloc((size_t)Mp*544*2);
  float* h_pre = (float*)alloc((size_t)N*1024*4);
  float* accR  = (float*)alloc((size_t)N*1024*4);
  float* xp    = (float*)alloc((size_t)N*512*4);
  float* el    = (float*)alloc((size_t)N*4*4);
  float* er    = (float*)alloc((size_t)N*4*4);
  float* ev    = (float*)alloc((size_t)E*4*4);
  float* mx    = (float*)alloc((size_t)N*4*4);
  float* ssum  = (float*)alloc((size_t)N*4*4);
  float* deg   = (float*)alloc((size_t)N*4);
  float* efacc = (float*)alloc((size_t)N*3*4);
  float* ef    = (float*)alloc((size_t)E*3*4);
  float* gate1 = (float*)alloc((size_t)N*128*4);
  float* gate  = (float*)alloc((size_t)N*4);
  float* wv    = (float*)alloc((size_t)N*4);
  float* gmax  = (float*)alloc((size_t)G*4);
  float* gsum  = (float*)alloc((size_t)G*4);
  float* emb   = (float*)alloc((size_t)G*1024*4);
  bf16* resPack[3], *fcPack[3], *npPack[3] = {nullptr, nullptr, nullptr};
  for (int li = 0; li < 3; ++li){
    resPack[li] = (bf16*)alloc((size_t)KresA[li]*FA[li]*2);
    fcPack[li]  = (bf16*)alloc((size_t)KfcA[li]*FA[li]*2);
    if (li > 0) npPack[li] = (bf16*)alloc((size_t)finA[li]*finA[li]*2);
  }
  bf16* poolPack = (bf16*)alloc((size_t)1024*128*2);

  float* outp  = (float*)d_out;            // [mu(G*64) | logvar(G*64) | h(N*1024)]
  float* h_out = outp + (size_t)2*G*64;

  // ---- setup: zero padded activation pool, convert inputs, pack weights ----
  fill_f32<<<nblk((long long)Mp*512), TPB, 0, stream>>>((float*)hbf, 0.f, (long long)Mp*512);
  convert_pad_bf16<<<nblk((long long)Mp*32), TPB, 0, stream>>>(node_feats, xbf, N, Mp, 4, 32);
  for (int li = 0; li < 3; ++li){
    pack_weight<<<nblk((long long)KresA[li]*FA[li]), TPB, 0, stream>>>(P(li,10), resPack[li], finA[li], KresA[li], FA[li]);
    pack_weight<<<nblk((long long)KfcA[li]*FA[li]),  TPB, 0, stream>>>(P(li,6),  fcPack[li],  finA[li]+3, KfcA[li], FA[li]);
    if (li > 0)
      pack_weight<<<nblk((long long)finA[li]*finA[li]), TPB, 0, stream>>>(P(li,4), npPack[li], finA[li], finA[li], finA[li]);
  }
  pack_weight<<<nblk((long long)1024*128), TPB, 0, stream>>>(pool_W1, poolPack, 1024, 1024, 128);

  // ---- GAT layers ----
  for (int li = 0; li < 3; ++li){
    const int fin = finA[li], fout = foutA[li], F = FA[li];
    const int Kres = KresA[li], Kfc = KfcA[li];

    // edge MLP and degree / edge-feature scatter
    edge_mlp_kernel<<<nblk(E), TPB, 0, stream>>>(edge_feats, P(li,0), P(li,1), P(li,2), P(li,3), ef, E);
    fill_f32<<<nblk(N), TPB, 0, stream>>>(deg, 0.f, N);
    fill_f32<<<nblk((long long)N*3), TPB, 0, stream>>>(efacc, 0.f, (long long)N*3);
    deg_ef_scatter<<<nblk(E), TPB, 0, stream>>>(ef, src, dst, deg, efacc, E);

    // xp = x @ np_W + np_b
    if (li == 0){
      xp0_kernel<<<nblk((long long)N*4), TPB, 0, stream>>>(node_feats, P(0,4), P(0,5), xp, N);
    } else {
      dim3 g(((fin>>4)+15)/16, Mp/32);
      gemm_wmma_bf16<<<g, TPB, 0, stream>>>(hbf, npPack[li], xp, P(li,5), N, fin, fin, 1);
    }

    // residual = x @ res_W + res_b  (accumulated into accR; GAT output adds on top)
    {
      dim3 g(((F>>4)+15)/16, Mp/32);
      const bf16* A = (li == 0) ? xbf : hbf;
      gemm_wmma_bf16<<<g, TPB, 0, stream>>>(A, resPack[li], accR, P(li,11), N, Kres, F, 1);
    }

    // temp (bf16, padded) and h = temp @ fc_W
    build_tempb<<<nblk((long long)Mp*Kfc), TPB, 0, stream>>>(xp, deg, efacc, tempb, N, Mp, fin, Kfc);
    {
      dim3 g(((F>>4)+15)/16, Mp/32);
      gemm_wmma_bf16<<<g, TPB, 0, stream>>>(tempb, fcPack[li], h_pre, nullptr, N, Kfc, F, 0);
    }

    // attention logits + edge softmax grouped by dst
    el_er_kernel<<<nblk((long long)N*4), TPB, 0, stream>>>(h_pre, P(li,7), P(li,8), el, er, N, F, fout);
    fill_f32<<<nblk((long long)N*4), TPB, 0, stream>>>(mx, -3.0e38f, (long long)N*4);
    fill_f32<<<nblk((long long)N*4), TPB, 0, stream>>>(ssum, 0.f, (long long)N*4);
    edge_logits<<<nblk((long long)E*4), TPB, 0, stream>>>(el, er, src, dst, ev, mx, E);
    edge_exp   <<<nblk((long long)E*4), TPB, 0, stream>>>(ev, dst, mx, ssum, E);
    edge_norm  <<<nblk((long long)E*4), TPB, 0, stream>>>(ev, dst, ssum, E);

    // gat_out scatter-add into residual buffer
    long long tot = (long long)E * F;
    gat_scatter<<<nblk(tot), TPB, 0, stream>>>(h_pre, ev, src, dst, accR, tot, F, fout);

    // LayerNorm(gat+res) -> SiLU ; write bf16 for next layer, fp32 h on last layer
    float* of32 = (li == 2) ? h_out : nullptr;
    ln_silu_kernel<<<N, TPB, 0, stream>>>(accR, P(li,9), P(li,12), P(li,13), of32, hbf, F);
  }

  // ---- GlobalAttentionPooling + VAE heads ----
  {
    dim3 g(((128>>4)+15)/16, Mp/32);
    gemm_wmma_bf16<<<g, TPB, 0, stream>>>(hbf, poolPack, gate1, pool_b1, N, 1024, 128, 1);
  }
  pool_gate_kernel<<<nblk(N), TPB, 0, stream>>>(gate1, pool_W2, pool_b2, gate, N);
  fill_f32<<<1, TPB, 0, stream>>>(gmax, -3.0e38f, G);
  fill_f32<<<1, TPB, 0, stream>>>(gsum, 0.f, G);
  fill_f32<<<nblk((long long)G*1024), TPB, 0, stream>>>(emb, 0.f, (long long)G*1024);
  graph_max   <<<nblk(N), TPB, 0, stream>>>(gate, gid, gmax, N);
  graph_expsum<<<nblk(N), TPB, 0, stream>>>(gate, gid, gmax, wv, gsum, N);
  graph_emb_scatter<<<nblk((long long)N*1024), TPB, 0, stream>>>(h_out, wv, gsum, gid, emb, N);
  mu_lv_kernel<<<nblk((long long)G*64), TPB, 0, stream>>>(emb, mu_W, mu_b, lv_W, lv_b, outp, G);
}
